// IntraModalityAttention_12223476924497
// MI455X (gfx1250) — matrix operand order repaired and checked by reading, not compiled, not run on hardware
//
#include <hip/hip_runtime.h>

// ---------------------------------------------------------------------------
// IntraModalityAttention for MI455X (gfx1250, wave32, WMMA).
// Split-f16 (hi/lo) GEMMs on v_wmma_f32_16x16x32_f16 (3 products per tile ->
// ~f32 accuracy at ~3x f16 cost, still ~8/3x faster than f32 WMMA and near
// the 23.3 TB/s memory floor of ~15us for ~340MB of traffic).
// ---------------------------------------------------------------------------

typedef _Float16 f16;
typedef __attribute__((ext_vector_type(16))) _Float16 v16h;
typedef __attribute__((ext_vector_type(8)))  float    v8f;
typedef unsigned int   u32;
typedef unsigned short u16;

#define IN_C  1024
#define MIDC  256
#define S_C   256
#define BATCH 32
#define HW    784          // 28*28
#define NT16  49           // 784 / 16
#define KTOP  196          // int(0.25 * 784)

union FragCast { u32 u[8]; v16h v; };

// A-matrix 16x32 f16 fragment (ISA 7.12.2): lanes 0-15 row M=lane, K in
// {0..7,16..23}; lanes 16-31 K in {8..15,24..31}. q = uint ptr at row start
// (row*pitch + k0, both even). Each uint = 2 consecutive-K f16.
__device__ __forceinline__ v16h load_frag_a(const u32* q, int hlf) {
  FragCast f; const int s = hlf * 4;
  f.u[0] = q[s + 0]; f.u[1] = q[s + 1]; f.u[2] = q[s + 2];  f.u[3] = q[s + 3];
  f.u[4] = q[s + 8]; f.u[5] = q[s + 9]; f.u[6] = q[s + 10]; f.u[7] = q[s + 11];
  return f.v;
}
// B-matrix 32x16 f16 fragment: lane n = lane&15; lanes 0-15 hold K in
// {0..7,16..23}, lanes 16-31 hold K in {8..15,24..31} (same order as A).
__device__ __forceinline__ v16h load_frag_b(const u32* q, int hlf) {
  FragCast f; const int s = hlf * 4;
#pragma unroll
  for (int j = 0; j < 8; ++j) f.u[j] = q[s + (j & 3) + (j >> 2) * 8];
  return f.v;
}

__device__ __forceinline__ v8f wmma(v16h a, v16h b, v8f c) {
  // 8 args: (neg_a, A, neg_b, B, c_mod, C, reuse_a, reuse_b)
  return __builtin_amdgcn_wmma_f32_16x16x32_f16(false, a, false, b,
                                                (short)0, c, false, false);
}

__device__ __forceinline__ void split16(float v, u16& h, u16& l) {
  f16 hi = (f16)v;
  float hf = (float)hi;
  f16 lo = (f16)(v - hf);
  h = __builtin_bit_cast(u16, hi);
  l = __builtin_bit_cast(u16, lo);
}

// --------------------------- prep kernels ----------------------------------

__global__ __launch_bounds__(256) void prep_split_kernel(
    const float* __restrict__ wq, const float* __restrict__ ww,
    u16* __restrict__ wq_hi, u16* __restrict__ wq_lo,
    u16* __restrict__ ww_hi, u16* __restrict__ ww_lo) {
  const int i = blockIdx.x * 256 + threadIdx.x;   // exactly 262144 threads
  u16 h, l;
  split16(wq[i], h, l); wq_hi[i] = h; wq_lo[i] = l;
  split16(ww[i], h, l); ww_hi[i] = h; ww_lo[i] = l;
}

__global__ __launch_bounds__(256) void prep_us_kernel(
    const float* __restrict__ s, const float* __restrict__ uq,
    const float* __restrict__ wqb, float* __restrict__ us) {
  const int m = threadIdx.x, b = blockIdx.x;
  float acc = wqb[m];
  const float* sr = s + b * S_C;
  const float* ur = uq + m * S_C;
#pragma unroll 4
  for (int k = 0; k < S_C; ++k) acc += ur[k] * sr[k];
  us[b * MIDC + m] = acc;
}

// --------------- GEMM1: sig = stanh(Wq*x + b + Uq*s) -----------------------
// grid (25 n-pairs, 32 b), 256 threads (8 waves). Wave w owns M-subtiles
// w*16 and 128+w*16; block covers 32 spatial columns.

__global__ __launch_bounds__(256) void gemm1_stanh_kernel(
    const float* __restrict__ x,
    const u16* __restrict__ wq_hi, const u16* __restrict__ wq_lo,
    const float* __restrict__ us,
    u16* __restrict__ sig_hi, u16* __restrict__ sig_lo) {
  __shared__ u16 bH[32 * 32];   // x tile split-hi, [n][k] (k contiguous)
  __shared__ u16 bL[32 * 32];   // x tile split-lo
  const int pair = blockIdx.x, b = blockIdx.y;
  const int n0 = pair * 32;
  const int t = threadIdx.x;
  const int lane = t & 31, wave = t >> 5;
  const int hlf = lane >> 4, lrow = lane & 15;
  const int m0a = wave * 16, m0b = 128 + wave * 16;

  v8f acc00 = {0.f, 0.f, 0.f, 0.f, 0.f, 0.f, 0.f, 0.f};
  v8f acc01 = acc00, acc10 = acc00, acc11 = acc00;

  for (int k0 = 0; k0 < IN_C; k0 += 32) {
    __syncthreads();
    // stage x[k0..k0+31][n0..n0+31] transposed+split into LDS
#pragma unroll
    for (int ii = 0; ii < 4; ++ii) {
      const int i = t + ii * 256;
      const int k = i >> 5, n = i & 31;
      const int gn = n0 + n;
      float v = (gn < HW) ? x[((size_t)b * IN_C + (k0 + k)) * HW + gn] : 0.f;
      u16 h, l; split16(v, h, l);
      bH[n * 32 + k] = h;
      bL[n * 32 + k] = l;
    }
    __syncthreads();

    v16h A0h = load_frag_a((const u32*)(wq_hi + (size_t)(m0a + lrow) * IN_C + k0), hlf);
    v16h A0l = load_frag_a((const u32*)(wq_lo + (size_t)(m0a + lrow) * IN_C + k0), hlf);
    v16h A1h = load_frag_a((const u32*)(wq_hi + (size_t)(m0b + lrow) * IN_C + k0), hlf);
    v16h A1l = load_frag_a((const u32*)(wq_lo + (size_t)(m0b + lrow) * IN_C + k0), hlf);
    v16h B0h = load_frag_b((const u32*)(bH + lrow * 32), hlf);
    v16h B0l = load_frag_b((const u32*)(bL + lrow * 32), hlf);
    v16h B1h = load_frag_b((const u32*)(bH + (16 + lrow) * 32), hlf);
    v16h B1l = load_frag_b((const u32*)(bL + (16 + lrow) * 32), hlf);

    acc00 = wmma(A0h, B0h, acc00); acc00 = wmma(A0h, B0l, acc00); acc00 = wmma(A0l, B0h, acc00);
    acc01 = wmma(A0h, B1h, acc01); acc01 = wmma(A0h, B1l, acc01); acc01 = wmma(A0l, B1h, acc01);
    acc10 = wmma(A1h, B0h, acc10); acc10 = wmma(A1h, B0l, acc10); acc10 = wmma(A1l, B0h, acc10);
    acc11 = wmma(A1h, B1h, acc11); acc11 = wmma(A1h, B1l, acc11); acc11 = wmma(A1l, B1h, acc11);
  }

  // epilogue: +us, scaled tanh, split to hi/lo f16, store sig[b][n][m]
  // (m contiguous -> one 16B store each for hi and lo per lane per tile)
  auto store_sig = [&](const v8f& acc, int m0, int nsub) {
    const int n = n0 + nsub * 16 + lrow;
    if (n >= HW) return;
    const int mbase = m0 + hlf * 8;
    u32 hws[4], lws[4];
#pragma unroll
    for (int r = 0; r < 8; ++r) {
      float v = acc[r] + us[b * MIDC + mbase + r];
      v = 1.7159f * tanhf(0.66666667f * v);
      u16 h, l; split16(v, h, l);
      if (r & 1) { hws[r >> 1] |= ((u32)h) << 16; lws[r >> 1] |= ((u32)l) << 16; }
      else       { hws[r >> 1]  = (u32)h;         lws[r >> 1]  = (u32)l;         }
    }
    const size_t off = ((size_t)b * HW + n) * MIDC + mbase;
    uint4 hv; hv.x = hws[0]; hv.y = hws[1]; hv.z = hws[2]; hv.w = hws[3];
    uint4 lv; lv.x = lws[0]; lv.y = lws[1]; lv.z = lws[2]; lv.w = lws[3];
    *reinterpret_cast<uint4*>(sig_hi + off) = hv;
    *reinterpret_cast<uint4*>(sig_lo + off) = lv;
  };
  store_sig(acc00, m0a, 0); store_sig(acc01, m0a, 1);
  store_sig(acc10, m0b, 0); store_sig(acc11, m0b, 1);
}

// ------- GEMM2 + top-k threshold + softmax + weighted reduce ---------------
// grid (64 c-tiles, 32 b), 256 threads. Phase 1: waves stripe the 49 n-tiles,
// alpha rows land in LDS. Phase 2: wave w handles rows w and w+8.

#define APITCH 788   // 784 + 4 pad (bank stagger)

__global__ __launch_bounds__(256) void gemm2_topk_softmax_kernel(
    const float* __restrict__ x,
    const u16* __restrict__ ww_hi, const u16* __restrict__ ww_lo,
    const u16* __restrict__ sig_hi, const u16* __restrict__ sig_lo,
    const float* __restrict__ wb,
    float* __restrict__ out_c, float* __restrict__ out_atten) {
  __shared__ float alphaS[16 * APITCH];   // ~50 KB
  const int c0 = blockIdx.x * 16, b = blockIdx.y;
  const int t = threadIdx.x;
  const int lane = t & 31, wave = t >> 5;
  const int hlf = lane >> 4, lrow = lane & 15;

  // ---- phase 1: alpha = ww * sig + wb, rows c0..c0+15, all 784 columns ----
  const u32* qah = (const u32*)(ww_hi + (size_t)(c0 + lrow) * MIDC);
  const u32* qal = (const u32*)(ww_lo + (size_t)(c0 + lrow) * MIDC);
  for (int nt = wave; nt < NT16; nt += 8) {      // wave-uniform loop
    const int n0 = nt * 16;
    v8f acc = {0.f, 0.f, 0.f, 0.f, 0.f, 0.f, 0.f, 0.f};
    const u32* qbh = (const u32*)(sig_hi + ((size_t)b * HW + n0 + lrow) * MIDC);
    const u32* qbl = (const u32*)(sig_lo + ((size_t)b * HW + n0 + lrow) * MIDC);
#pragma unroll
    for (int kb = 0; kb < 8; ++kb) {             // K = 256, 32-deep steps
      const int ku = kb * 16;                    // 32 halfwords = 16 uints
      v16h Ah = load_frag_a(qah + ku, hlf);
      v16h Al = load_frag_a(qal + ku, hlf);
      v16h Bh = load_frag_b(qbh + ku, hlf);
      v16h Bl = load_frag_b(qbl + ku, hlf);
      acc = wmma(Ah, Bh, acc); acc = wmma(Ah, Bl, acc); acc = wmma(Al, Bh, acc);
    }
    const int n = n0 + lrow;
#pragma unroll
    for (int r = 0; r < 8; ++r) {
      const int crow = r + hlf * 8;
      alphaS[crow * APITCH + n] = acc[r] + wb[c0 + crow];
    }
  }
  __syncthreads();

  // ---- phase 2: per-row exact top-k threshold, softmax, reduce ----
#pragma unroll
  for (int rr = 0; rr < 2; ++rr) {
    const int crow = wave + rr * 8;
    const int c = c0 + crow;
    float vals[25]; u32 uv[25];
    float mx = -3.402823466e38f;
#pragma unroll
    for (int j = 0; j < 25; ++j) {
      const int idx = lane + j * 32;
      float v = 0.f; u32 u = 0u;                 // sentinel u=0 never counts
      if (idx < HW) {
        v = alphaS[crow * APITCH + idx];
        const u32 bu = __builtin_bit_cast(u32, v);
        u = (bu & 0x80000000u) ? ~bu : (bu | 0x80000000u);  // order-preserving
        mx = fmaxf(mx, v);
      }
      vals[j] = v; uv[j] = u;
    }
#pragma unroll
    for (int o = 16; o > 0; o >>= 1) mx = fmaxf(mx, __shfl_xor(mx, o, 32));
    const float M = fmaxf(mx, 0.f);              // zeros join the softmax

    // kth-largest (k=196): largest th with count(u >= th) >= k; ties kept.
    u32 res = 0u;
    for (int bit = 31; bit >= 0; --bit) {
      const u32 cand = res | (1u << bit);
      int cnt = 0;
#pragma unroll
      for (int j = 0; j < 25; ++j) cnt += (uv[j] >= cand) ? 1 : 0;
#pragma unroll
      for (int o = 16; o > 0; o >>= 1) cnt += __shfl_xor(cnt, o, 32);
      if (cnt >= KTOP) res = cand;
    }

    float ssum = 0.f;
#pragma unroll
    for (int j = 0; j < 25; ++j) {
      const int idx = lane + j * 32;
      if (idx < HW) {
        const float v2 = (uv[j] < res) ? 0.f : vals[j];   // strict < zeroed
        const float e = __expf(v2 - M);
        vals[j] = e; ssum += e;
      }
    }
#pragma unroll
    for (int o = 16; o > 0; o >>= 1) ssum += __shfl_xor(ssum, o, 32);
    const float inv = 1.f / ssum;

    const float* xr = x + ((size_t)b * IN_C + c) * HW;
    float* oa = out_atten + ((size_t)b * IN_C + c) * HW;
    float dacc = 0.f;
#pragma unroll
    for (int j = 0; j < 25; ++j) {
      const int idx = lane + j * 32;
      if (idx < HW) {
        const float a = vals[j] * inv;
        oa[idx] = a;                              // coalesced
        dacc += xr[idx] * a;
      }
    }
#pragma unroll
    for (int o = 16; o > 0; o >>= 1) dacc += __shfl_xor(dacc, o, 32);
    if (lane == 0) out_c[b * IN_C + c] = dacc;
  }
}

// ------------------------------- launch ------------------------------------

extern "C" void kernel_launch(void* const* d_in, const int* in_sizes, int n_in,
                              void* d_out, int out_size, void* d_ws, size_t ws_size,
                              hipStream_t stream) {
  (void)in_sizes; (void)n_in; (void)out_size; (void)ws_size;
  const float* x   = (const float*)d_in[0];
  const float* s   = (const float*)d_in[1];
  const float* wq  = (const float*)d_in[2];
  const float* wqb = (const float*)d_in[3];
  const float* uq  = (const float*)d_in[4];
  const float* ww  = (const float*)d_in[5];
  const float* wb  = (const float*)d_in[6];

  // workspace layout (~27.8 MB total)
  char* ws = (char*)d_ws;
  u16*   wq_hi  = (u16*)(ws + 0);                         // 512 KB
  u16*   wq_lo  = (u16*)(ws + 524288);                    // 512 KB
  u16*   ww_hi  = (u16*)(ws + 1048576);                   // 512 KB
  u16*   ww_lo  = (u16*)(ws + 1572864);                   // 512 KB
  float* us     = (float*)(ws + 2097152);                 // 32 KB
  u16*   sig_hi = (u16*)(ws + 2129920);                   // 12.845 MB
  u16*   sig_lo = (u16*)(ws + 2129920 + 12845056);        // 12.845 MB

  float* out_c     = (float*)d_out;                       // [32,1024]
  float* out_atten = out_c + BATCH * IN_C;                // [32,1024,784]

  prep_split_kernel<<<1024, 256, 0, stream>>>(wq, ww, wq_hi, wq_lo, ww_hi, ww_lo);
  prep_us_kernel<<<BATCH, 256, 0, stream>>>(s, uq, wqb, us);
  gemm1_stanh_kernel<<<dim3(25, BATCH), 256, 0, stream>>>(x, wq_hi, wq_lo, us,
                                                          sig_hi, sig_lo);
  gemm2_topk_softmax_kernel<<<dim3(IN_C / 16, BATCH), 256, 0, stream>>>(
      x, ww_hi, ww_lo, sig_hi, sig_lo, wb, out_c, out_atten);
}
